// MultiTimeframeHybridTIN_25434796327340
// MI455X (gfx1250) — matrix operand order repaired and checked by reading, not compiled, not run on hardware
//
#include <hip/hip_runtime.h>
#include <hip/hip_bf16.h>

// ---------------------------------------------------------------------------
// Problem constants
// ---------------------------------------------------------------------------
#define BB   256   // batch
#define SS   512   // sequence length
#define HH   64    // hidden
#define GQ   256   // 4*H gate width
#define INP  32    // padded input feature count (9 sigs + 10 ctx -> pad to 32)
#define TCH  8     // timesteps per TDM chunk

typedef __attribute__((ext_vector_type(16))) _Float16 v16h;
typedef __attribute__((ext_vector_type(8)))  _Float16 v8h;
typedef __attribute__((ext_vector_type(8)))  float    v8f;
typedef __attribute__((ext_vector_type(4)))  unsigned int u32x4;
typedef __attribute__((ext_vector_type(8)))  int      i32x8;
typedef __attribute__((ext_vector_type(4)))  int      i32x4;

#if defined(__has_builtin)
#if __has_builtin(__builtin_amdgcn_tensor_load_to_lds) && \
    __has_builtin(__builtin_amdgcn_tensor_store_from_lds) && \
    __has_builtin(__builtin_amdgcn_s_wait_tensorcnt)
#define USE_TDM 1
#endif
#endif
#ifndef USE_TDM
#define USE_TDM 0
#endif

// ---------------------------------------------------------------------------
// Input index map (JAX tree-leaves order: dicts sorted by key, lists in order)
// ---------------------------------------------------------------------------
enum {
  IN_CONTEXT = 0,
  IN_OHLC    = 1,
  IN_ADV_W1  = 2, IN_ADV_W2 = 3, IN_ADV_B1 = 4, IN_ADV_B2 = 5,
  IN_ATR_W   = 6, IN_BB_K   = 7, IN_BB_W   = 8,
  IN_L0_WHH  = 9, IN_L0_WIH = 10, IN_L0_BHH = 11, IN_L0_BIH = 12,
  IN_L1_WHH  = 13, IN_L1_WIH = 14, IN_L1_BHH = 15, IN_L1_BIH = 16,
  IN_M1F_FAST = 17, IN_M1F_SIG = 18, IN_M1F_SLOW = 19,
  IN_M1S_FAST = 20, IN_M1S_SIG = 21, IN_M1S_SLOW = 22,
  IN_M5F_FAST = 23, IN_M5F_SIG = 24, IN_M5F_SLOW = 25,
  IN_M5S_FAST = 26, IN_M5S_SIG = 27, IN_M5S_SLOW = 28,
  IN_RSI_GAIN = 29, IN_RSI_LOSS = 30,
  IN_VAL_W1 = 31, IN_VAL_W2 = 32, IN_VAL_B1 = 33, IN_VAL_B2 = 34,
  IN_P15 = 35, IN_P1H = 36, IN_P5 = 37
};

// ---------------------------------------------------------------------------
// Workspace layout (bytes)
// ---------------------------------------------------------------------------
#define OFF_FEATS  0
#define OFF_HSEQ   (OFF_FEATS + BB*SS*INP*2)
#define OFF_HLAST  (OFF_HSEQ  + BB*SS*HH*2)
#define OFF_WIH0   (OFF_HLAST + BB*HH*2)
#define OFF_WHH0   (OFF_WIH0  + GQ*INP*2)
#define OFF_WIH1   (OFF_WHH0  + GQ*HH*2)
#define OFF_WHH1   (OFF_WIH1  + GQ*HH*2)
#define OFF_B0     (OFF_WHH1  + GQ*HH*2)
#define OFF_B1     (OFF_B0    + GQ*4)

// ---------------------------------------------------------------------------
// Fragment loaders (contiguous k = 16h + s in A and B, wave32)
// ---------------------------------------------------------------------------
__device__ __forceinline__ v16h load_afrag(const _Float16* row, int k0, int lane) {
  int kb = (lane & 16) ? 16 : 0;
  v8h lo = *(const v8h*)(row + k0 + kb);
  v8h hi = *(const v8h*)(row + k0 + kb + 8);
  return __builtin_shufflevector(lo, hi, 0,1,2,3,4,5,6,7,8,9,10,11,12,13,14,15);
}

__device__ __forceinline__ v16h load_bfrag(const _Float16* W, int ldK, int col,
                                           int k0, int lane) {
  const _Float16* p = W + (size_t)col * ldK + k0 + ((lane & 16) ? 16 : 0);
  return *(const v16h*)p;
}

__device__ __forceinline__ float sigmoidf(float x) {
  return 1.0f / (1.0f + __expf(-x));
}
// branch-free tanh (avoids libm's EXEC-mask divergence in the inner loop)
__device__ __forceinline__ float tanh_fast(float x) {
  float e = __expf(2.0f * x);
  return 1.0f - 2.0f / (e + 1.0f);
}

// ---------------------------------------------------------------------------
// TDM: issue a 2D f16 tile copy (16 lines x tile_d0 elements, global line
// stride stride_el elements) between global and LDS.  D# per ISA ch.8.
// ---------------------------------------------------------------------------
#if USE_TDM
__device__ __forceinline__ void tdm_issue(bool is_store, unsigned lds_addr,
                                          const void* gptr, unsigned tile_d0,
                                          unsigned tile_d1,
                                          unsigned long long stride_el) {
  unsigned long long ga = (unsigned long long)(size_t)gptr;
  u32x4 g0 = { 1u,                                   // count=1 (valid, user)
               lds_addr,                             // lds_addr [63:32]
               (unsigned)(ga & 0xFFFFFFFFu),         // global_addr[31:0]
               (unsigned)((ga >> 32) & 0x01FFFFFFu)  // global_addr[56:32]
                   | (2u << 30) };                   // type=2 ("image")
  i32x8 g1 = { (int)(1u << 16),                      // data_size=1 (2 bytes)
               (int)((tile_d0 & 0xFFFFu) << 16),     // tensor_dim0[15:0]
               (int)(((tile_d0 >> 16) & 0xFFFFu) |   // tensor_dim0[31:16]
                     ((tile_d1 & 0xFFFFu) << 16)),   // tensor_dim1[15:0]
               (int)(((tile_d1 >> 16) & 0xFFFFu) |   // tensor_dim1[31:16]
                     ((tile_d0 & 0xFFFFu) << 16)),   // tile_dim0
               (int)(tile_d1 & 0xFFFFu),             // tile_dim1 (tile_dim2=0)
               (int)(stride_el & 0xFFFFFFFFull),     // tensor_dim0_stride lo
               (int)((stride_el >> 32) & 0xFFFFull), // tensor_dim0_stride hi
               0 };
  i32x4 z4 = { 0, 0, 0, 0 };
#if __clang_major__ >= 23
  i32x8 z8 = { 0, 0, 0, 0, 0, 0, 0, 0 };
  if (is_store) __builtin_amdgcn_tensor_store_from_lds(g0, g1, z4, z4, z8, 0);
  else          __builtin_amdgcn_tensor_load_to_lds(g0, g1, z4, z4, z8, 0);
#else
  if (is_store) __builtin_amdgcn_tensor_store_from_lds(g0, g1, z4, z4, 0);
  else          __builtin_amdgcn_tensor_load_to_lds(g0, g1, z4, z4, 0);
#endif
}
#endif

// ---------------------------------------------------------------------------
// Kernel 1: weight prep (f32 -> f16, pad Wih0 19->32, fuse biases)
// ---------------------------------------------------------------------------
__global__ __launch_bounds__(256) void prep_weights(
    const float* __restrict__ Wih0, const float* __restrict__ Whh0,
    const float* __restrict__ bih0, const float* __restrict__ bhh0,
    const float* __restrict__ Wih1, const float* __restrict__ Whh1,
    const float* __restrict__ bih1, const float* __restrict__ bhh1,
    _Float16* __restrict__ wih0h, _Float16* __restrict__ whh0h,
    _Float16* __restrict__ wih1h, _Float16* __restrict__ whh1h,
    float* __restrict__ b0, float* __restrict__ b1) {
  int gid = blockIdx.x * 256 + threadIdx.x;
  if (gid < 8192) {
    int r = gid >> 5, c = gid & 31;
    wih0h[gid] = (_Float16)((c < 19) ? Wih0[r * 19 + c] : 0.0f);
  } else if (gid < 24576) {
    int i = gid - 8192;  whh0h[i] = (_Float16)Whh0[i];
  } else if (gid < 40960) {
    int i = gid - 24576; wih1h[i] = (_Float16)Wih1[i];
  } else if (gid < 57344) {
    int i = gid - 40960; whh1h[i] = (_Float16)Whh1[i];
  } else if (gid < 57600) {
    int i = gid - 57344; b0[i] = bih0[i] + bhh0[i];
  } else if (gid < 57856) {
    int i = gid - 57600; b1[i] = bih1[i] + bhh1[i];
  }
}

// ---------------------------------------------------------------------------
// Kernel 2: technical-indicator features -> f16 [B,S,32]  (memory-bound)
// ---------------------------------------------------------------------------
__device__ float macd_sig(const float* __restrict__ x, int L,
                          const float* __restrict__ wf, int F,
                          const float* __restrict__ ws, int Sl,
                          const float* __restrict__ wsig, int G) {
  float hist_last = 0.0f, dot = 0.0f;
  for (int t = 0; t < G; ++t) {
    int end = L - G + 1 + t;
    float vf = 0.0f;
    for (int i = 0; i < F; ++i) vf += x[end - F + i] * wf[i];
    float vs = 0.0f;
    for (int i = 0; i < Sl; ++i) vs += x[end - Sl + i] * ws[i];
    float hv = vf - vs;
    dot += hv * wsig[t];
    if (t == G - 1) hist_last = hv;
  }
  return hist_last - dot;
}

__global__ __launch_bounds__(128) void features_kernel(
    const float* __restrict__ p5, const float* __restrict__ p15,
    const float* __restrict__ ohlc, const float* __restrict__ p1h,
    const float* __restrict__ ctx,
    const float* __restrict__ m5f_f, const float* __restrict__ m5f_s, const float* __restrict__ m5f_g,
    const float* __restrict__ m5s_f, const float* __restrict__ m5s_s, const float* __restrict__ m5s_g,
    const float* __restrict__ m1f_f, const float* __restrict__ m1f_s, const float* __restrict__ m1f_g,
    const float* __restrict__ m1s_f, const float* __restrict__ m1s_s, const float* __restrict__ m1s_g,
    const float* __restrict__ rsi_g, const float* __restrict__ rsi_l,
    const float* __restrict__ atr_w, const float* __restrict__ bb_w,
    const float* __restrict__ bb_kk,
    _Float16* __restrict__ feats) {
  int gid = blockIdx.x * 128 + threadIdx.x;
  const float* x5 = p5 + (size_t)gid * 128;
  const float* x15 = p15 + (size_t)gid * 64;
  const float* xo = ohlc + (size_t)gid * 256;
  const float* x1 = p1h + (size_t)gid * 128;

  float sig[9];
  sig[0] = macd_sig(x5, 128, m5f_f, 6, m5f_s, 13, m5f_g, 5);
  sig[1] = macd_sig(x5, 128, m5s_f, 24, m5s_s, 52, m5s_g, 18);
  { float cn = x5[118]; sig[2] = (x5[127] - cn) / (cn + 1e-8f); }
  { float cn = x5[106]; sig[3] = (x5[127] - cn) / (cn + 1e-8f); }
  {
    float ag = 0.0f, al = 0.0f;
    for (int j = 0; j < 14; ++j) {
      float d = x15[50 + j] - x15[49 + j];
      ag += fmaxf(d, 0.0f) * rsi_g[j];
      al += fmaxf(-d, 0.0f) * rsi_l[j];
    }
    float r = 100.0f - 100.0f / (1.0f + ag / (al + 1e-8f));
    sig[4] = r / 50.0f - 1.0f;
  }
  {
    float a = 0.0f;
    float lastclose = xo[63 * 4 + 3];
    for (int j = 0; j < 14; ++j) {
      int l = 50 + j;
      float hi = xo[l * 4 + 1], lo = xo[l * 4 + 2], pc = xo[(l - 1) * 4 + 3];
      float tr = fmaxf(fmaxf(hi - lo, fabsf(hi - pc)), fabsf(lo - pc));
      a += tr * atr_w[j];
    }
    sig[5] = a / (lastclose + 1e-8f);
  }
  {
    float ma = 0.0f, sum = 0.0f;
    for (int j = 0; j < 20; ++j) { float v = x15[44 + j]; ma += v * bb_w[j]; sum += v; }
    float mean = sum * (1.0f / 20.0f);
    float var = 0.0f;
    for (int j = 0; j < 20; ++j) { float d = x15[44 + j] - mean; var += d * d; }
    float sd = sqrtf(var * (1.0f / 19.0f));
    float k = bb_kk[0];
    float up = ma + k * sd, lo = ma - k * sd;
    float pb = (x15[63] - lo) / (up - lo + 1e-8f);
    sig[6] = pb * 2.0f - 1.0f;
  }
  sig[7] = macd_sig(x1, 128, m1f_f, 6, m1f_s, 13, m1f_g, 5);
  sig[8] = macd_sig(x1, 128, m1s_f, 24, m1s_s, 52, m1s_g, 18);

  _Float16* f = feats + (size_t)gid * INP;
  const float* c = ctx + (size_t)gid * 10;
  for (int j = 0; j < 9; ++j)  f[j] = (_Float16)sig[j];
  for (int j = 0; j < 10; ++j) f[9 + j] = (_Float16)c[j];
  for (int j = 19; j < 32; ++j) f[j] = (_Float16)0.0f;
}

// ---------------------------------------------------------------------------
// Kernel 3: one LSTM layer, WMMA f16->f32 with TDM-staged x operand.
// ---------------------------------------------------------------------------
template <int KX, bool STORE_ALL>
__global__ __launch_bounds__(256) void lstm_layer(
    const _Float16* __restrict__ X,     // [B,S,KX] f16
    const _Float16* __restrict__ Wih,   // [256,KX] f16 row-major
    const _Float16* __restrict__ Whh,   // [256,64] f16 row-major
    const float* __restrict__ bias,     // [256]
    _Float16* __restrict__ Hout) {      // STORE_ALL ? [B,S,64] : [B,64]
  __shared__ float    sG[16][GQ];
  __shared__ float    sC[16][HH];
  __shared__ _Float16 sH[16][HH];
  __shared__ float    sB[GQ];
#if USE_TDM
  __shared__ _Float16 sX[2][16 * TCH * KX];               // x double buffer
  __shared__ _Float16 sY[STORE_ALL ? (16 * TCH * HH) : 1];  // h-seq staging
#endif

  const int tid = threadIdx.x;
  const int wave = tid >> 5;
  const int lane = tid & 31;
  const int bglob = blockIdx.x * 16;
  const int mrow = lane & 15;

  for (int i = tid; i < 16 * HH; i += 256) {
    (&sC[0][0])[i] = 0.0f;
    (&sH[0][0])[i] = (_Float16)0.0f;
  }
  sB[tid] = bias[tid];
  __syncthreads();

  // --- time-invariant weight B-fragments in registers ---
  const int ct0 = wave * 2;
  v16h bx[2][KX / 32];
  v16h bh[2][2];
#pragma unroll
  for (int cc = 0; cc < 2; ++cc) {
    int col = (ct0 + cc) * 16 + mrow;
#pragma unroll
    for (int kc = 0; kc < KX / 32; ++kc) bx[cc][kc] = load_bfrag(Wih, KX, col, kc * 32, lane);
#pragma unroll
    for (int kc = 0; kc < 2; ++kc)       bh[cc][kc] = load_bfrag(Whh, HH, col, kc * 32, lane);
  }

  const int erow = tid >> 4;
  const int ec0  = (tid & 15) * 4;

  auto matmul_step = [&](const _Float16* xrow) {
    v16h ax[KX / 32];
#pragma unroll
    for (int kc = 0; kc < KX / 32; ++kc) ax[kc] = load_afrag(xrow, kc * 32, lane);
    v16h ah[2];
#pragma unroll
    for (int kc = 0; kc < 2; ++kc) ah[kc] = load_afrag(&sH[mrow][0], kc * 32, lane);
#pragma unroll
    for (int cc = 0; cc < 2; ++cc) {
      v8f acc = {};
#pragma unroll
      for (int kc = 0; kc < KX / 32; ++kc)
        acc = __builtin_amdgcn_wmma_f32_16x16x32_f16(false, ax[kc], false, bx[cc][kc],
                                                     (short)0, acc, false, false);
#pragma unroll
      for (int kc = 0; kc < 2; ++kc)
        acc = __builtin_amdgcn_wmma_f32_16x16x32_f16(false, ah[kc], false, bh[cc][kc],
                                                     (short)0, acc, false, false);
      int n = (ct0 + cc) * 16 + mrow;
      int mb = (lane & 16) ? 8 : 0;
      float bv = sB[n];
#pragma unroll
      for (int r = 0; r < 8; ++r) sG[mb + r][n] = acc[r] + bv;
    }
  };

  auto gate_step = [&](int t, int tt) {
    (void)tt;
#pragma unroll
    for (int j = 0; j < 4; ++j) {
      int hc = ec0 + j;
      float gi = sG[erow][hc];
      float gf = sG[erow][64 + hc];
      float gg = sG[erow][128 + hc];
      float go = sG[erow][192 + hc];
      float c = sigmoidf(gf) * sC[erow][hc] + sigmoidf(gi) * tanh_fast(gg);
      float h = sigmoidf(go) * tanh_fast(c);
      sC[erow][hc] = c;
      _Float16 hh = (_Float16)h;
      sH[erow][hc] = hh;
      if (STORE_ALL) {
#if USE_TDM
        sY[(erow * TCH + tt) * HH + hc] = hh;
#else
        Hout[((size_t)(bglob + erow) * SS + t) * HH + hc] = hh;
#endif
      } else if (t == SS - 1) {
        Hout[(size_t)(bglob + erow) * HH + hc] = hh;
      }
    }
  };

#if USE_TDM
  const unsigned ldsX0 = (unsigned)(size_t)&sX[0][0];
  const unsigned ldsX1 = (unsigned)(size_t)&sX[1][0];
  if (wave == 0)
    tdm_issue(false, ldsX0, X + (size_t)bglob * SS * KX,
              TCH * KX, 16, (unsigned long long)SS * KX);

  for (int tc = 0; tc < SS / TCH; ++tc) {
    // wait: x chunk tc landed in LDS; previous h-seq tensor store drained
    if (wave == 0) __builtin_amdgcn_s_wait_tensorcnt(0);
    __syncthreads();
    if (wave == 0 && tc + 1 < SS / TCH)
      tdm_issue(false, ((tc + 1) & 1) ? ldsX1 : ldsX0,
                X + ((size_t)bglob * SS + (size_t)(tc + 1) * TCH) * KX,
                TCH * KX, 16, (unsigned long long)SS * KX);

    const _Float16* xb = &sX[tc & 1][0];
    for (int tt = 0; tt < TCH; ++tt) {
      const int t = tc * TCH + tt;
      matmul_step(xb + (size_t)(mrow * TCH + tt) * KX);
      __syncthreads();
      gate_step(t, tt);
      __syncthreads();
    }
    if (STORE_ALL && wave == 0)
      tdm_issue(true, (unsigned)(size_t)&sY[0],
                Hout + ((size_t)bglob * SS + (size_t)tc * TCH) * HH,
                TCH * HH, 16, (unsigned long long)SS * HH);
  }
  // s_endpgm performs an implicit wait-idle, draining the final tensor store.
#else
  const int arow = bglob + mrow;
  for (int t = 0; t < SS; ++t) {
    matmul_step(X + ((size_t)arow * SS + t) * KX);
    __syncthreads();
    if (t + 1 < SS)
      __builtin_prefetch(X + ((size_t)arow * SS + t + 1) * KX, 0, 3);
    gate_step(t, 0);
    __syncthreads();
  }
#endif
}

// ---------------------------------------------------------------------------
// Kernel 4: dueling head (tiny): out = val + adv - mean(adv)
// ---------------------------------------------------------------------------
__global__ __launch_bounds__(256) void head_kernel(
    const _Float16* __restrict__ Hlast,
    const float* __restrict__ aW1, const float* __restrict__ ab1,
    const float* __restrict__ aW2, const float* __restrict__ ab2,
    const float* __restrict__ vW1, const float* __restrict__ vb1,
    const float* __restrict__ vW2, const float* __restrict__ vb2,
    float* __restrict__ out) {
  int b = blockIdx.x * blockDim.x + threadIdx.x;
  if (b >= BB) return;
  float h[HH];
#pragma unroll
  for (int k = 0; k < HH; ++k) h[k] = (float)Hlast[b * HH + k];
  float adv0 = ab2[0], adv1 = ab2[1], adv2 = ab2[2];
  float val = vb2[0];
  for (int j = 0; j < 32; ++j) {
    float a = ab1[j], v = vb1[j];
#pragma unroll
    for (int k = 0; k < HH; ++k) {
      a += aW1[j * HH + k] * h[k];
      v += vW1[j * HH + k] * h[k];
    }
    a = fmaxf(a, 0.0f);
    v = fmaxf(v, 0.0f);
    adv0 += aW2[0 * 32 + j] * a;
    adv1 += aW2[1 * 32 + j] * a;
    adv2 += aW2[2 * 32 + j] * a;
    val  += vW2[j] * v;
  }
  float mean = (adv0 + adv1 + adv2) * (1.0f / 3.0f);
  out[b * 3 + 0] = val + adv0 - mean;
  out[b * 3 + 1] = val + adv1 - mean;
  out[b * 3 + 2] = val + adv2 - mean;
}

// ---------------------------------------------------------------------------
extern "C" void kernel_launch(void* const* d_in, const int* in_sizes, int n_in,
                              void* d_out, int out_size, void* d_ws, size_t ws_size,
                              hipStream_t stream) {
  const float* context = (const float*)d_in[IN_CONTEXT];
  const float* ohlc    = (const float*)d_in[IN_OHLC];
  const float* p15     = (const float*)d_in[IN_P15];
  const float* p1h     = (const float*)d_in[IN_P1H];
  const float* p5      = (const float*)d_in[IN_P5];

  char* ws = (char*)d_ws;
  _Float16* feats = (_Float16*)(ws + OFF_FEATS);
  _Float16* hseq  = (_Float16*)(ws + OFF_HSEQ);
  _Float16* hlast = (_Float16*)(ws + OFF_HLAST);
  _Float16* wih0h = (_Float16*)(ws + OFF_WIH0);
  _Float16* whh0h = (_Float16*)(ws + OFF_WHH0);
  _Float16* wih1h = (_Float16*)(ws + OFF_WIH1);
  _Float16* whh1h = (_Float16*)(ws + OFF_WHH1);
  float*    b0    = (float*)(ws + OFF_B0);
  float*    b1    = (float*)(ws + OFF_B1);

  prep_weights<<<227, 256, 0, stream>>>(
      (const float*)d_in[IN_L0_WIH], (const float*)d_in[IN_L0_WHH],
      (const float*)d_in[IN_L0_BIH], (const float*)d_in[IN_L0_BHH],
      (const float*)d_in[IN_L1_WIH], (const float*)d_in[IN_L1_WHH],
      (const float*)d_in[IN_L1_BIH], (const float*)d_in[IN_L1_BHH],
      wih0h, whh0h, wih1h, whh1h, b0, b1);

  features_kernel<<<(BB * SS) / 128, 128, 0, stream>>>(
      p5, p15, ohlc, p1h, context,
      (const float*)d_in[IN_M5F_FAST], (const float*)d_in[IN_M5F_SLOW], (const float*)d_in[IN_M5F_SIG],
      (const float*)d_in[IN_M5S_FAST], (const float*)d_in[IN_M5S_SLOW], (const float*)d_in[IN_M5S_SIG],
      (const float*)d_in[IN_M1F_FAST], (const float*)d_in[IN_M1F_SLOW], (const float*)d_in[IN_M1F_SIG],
      (const float*)d_in[IN_M1S_FAST], (const float*)d_in[IN_M1S_SLOW], (const float*)d_in[IN_M1S_SIG],
      (const float*)d_in[IN_RSI_GAIN], (const float*)d_in[IN_RSI_LOSS],
      (const float*)d_in[IN_ATR_W], (const float*)d_in[IN_BB_W], (const float*)d_in[IN_BB_K],
      feats);

  lstm_layer<INP, true><<<BB / 16, 256, 0, stream>>>(feats, wih0h, whh0h, b0, hseq);
  lstm_layer<HH, false><<<BB / 16, 256, 0, stream>>>(hseq, wih1h, whh1h, b1, hlast);

  head_kernel<<<1, 256, 0, stream>>>(
      hlast,
      (const float*)d_in[IN_ADV_W1], (const float*)d_in[IN_ADV_B1],
      (const float*)d_in[IN_ADV_W2], (const float*)d_in[IN_ADV_B2],
      (const float*)d_in[IN_VAL_W1], (const float*)d_in[IN_VAL_B1],
      (const float*)d_in[IN_VAL_W2], (const float*)d_in[IN_VAL_B2],
      (float*)d_out);
}
